// MultiHeadSelfAttention_18940805776017
// MI455X (gfx1250) — matrix operand order repaired and checked by reading, not compiled, not run on hardware
//
#include <hip/hip_runtime.h>
#include <stdint.h>

// MI455X / gfx1250, wave32. All matmuls via v_wmma_f32_16x16x32_bf16.
// GEMMs and flash attention both use TDM (tensor_load_to_lds) double-buffered
// LDS staging with TENSORcnt sync and ds_load_b128 fragment reads.

#define DIM   1024
#define HEADS 16
#define HD    64
#define SEQ   2048
#define BATCH 4
#define NTOK  (BATCH * SEQ)          // 8192
#define SCALE 0.03125f               // 1024^-0.5

typedef __attribute__((ext_vector_type(16))) __bf16    v16bf;
typedef __attribute__((ext_vector_type(8)))  float     v8f;
typedef __attribute__((ext_vector_type(4)))  uint32_t  v4u;
typedef __attribute__((ext_vector_type(4)))  int       v4i;
typedef __attribute__((ext_vector_type(8)))  int       v8i;

union AFrag {
    v16bf    v;
    uint32_t u[8];
    v4u      q[2];
};

__device__ inline uint16_t f2bf_bits(float f) {
    __bf16 b = (__bf16)f;
    return __builtin_bit_cast(unsigned short, b);
}

__device__ inline v8f bfwmma(const AFrag& a, const AFrag& b, v8f c) {
    return __builtin_amdgcn_wmma_f32_16x16x32_bf16(
        /*neg_a=*/false, a.v, /*neg_b=*/false, b.v,
        /*c_mod=*/(short)0, c, /*reuse_a=*/false, /*reuse_b=*/false);
}

// ---------------------------------------------------------------- TDM
// 2D bf16 tile load: tile1 rows x tile0 cols, row stride 'stride' (elements),
// packed row-major into LDS at byte offset lds_off. D# per ISA 8.3/8.4.
__device__ inline void tdm_load_2d(uint32_t lds_off, const uint16_t* gptr,
                                   uint32_t tile0, uint32_t tile1,
                                   uint32_t stride) {
    uint64_t ga = (uint64_t)(uintptr_t)gptr;
    v4u g0;
    g0[0] = 1u;                                   // count=1, user mode, no gather
    g0[1] = lds_off;                              // lds_addr
    g0[2] = (uint32_t)(ga & 0xffffffffu);         // global_addr[31:0]
    g0[3] = (uint32_t)((ga >> 32) & 0x01ffffffu)  // global_addr[56:32]
            | (2u << 30);                         // type=2 (image)
    uint32_t dim0 = tile0, dim1 = tile1;
    v8i g1;
    g1[0] = (int)(1u << 16);                      // data_size=1 => 2 bytes
    g1[1] = (int)((dim0 & 0xffffu) << 16);        // tensor_dim0[15:0]
    g1[2] = (int)((dim0 >> 16) | ((dim1 & 0xffffu) << 16));
    g1[3] = (int)((dim1 >> 16) | (tile0 << 16));  // tile_dim0
    g1[4] = (int)tile1;                           // tile_dim1, tile_dim2=0
    g1[5] = (int)stride;                          // tensor_dim0_stride[31:0]
    g1[6] = 0;                                    // stride0[47:32], stride1 lo
    g1[7] = 0;
    v4i z4 = {0, 0, 0, 0};
#if __clang_major__ >= 23
    v8i z8 = {0, 0, 0, 0, 0, 0, 0, 0};
    __builtin_amdgcn_tensor_load_to_lds(g0, g1, z4, z4, z8, 0);
#else
    __builtin_amdgcn_tensor_load_to_lds(g0, g1, z4, z4, 0);
#endif
}

__device__ inline uint32_t lds_off_u32(const void* p) {
    return (uint32_t)(uintptr_t)p;   // LDS flat addr: low 32 bits = LDS offset
}

// ---------------------------------------------------------------- convert
__global__ void cvt_f32_bf16(const float* __restrict__ in,
                             uint16_t* __restrict__ out, int n) {
    int i = blockIdx.x * blockDim.x + threadIdx.x;
    if (i < n) out[i] = f2bf_bits(in[i]);
}

// ------------------------------------------------- TDM-staged GEMM core
// WG = 256 thr / 8 waves (2x4), wave tile 64x64, WG tile 128x256, kstep 32.
// A-frag from LDS: two b128 at row*32 + {16g, 16g+8}. B-frag: two b128 at
// row*32 + 16g (+8). Wave 0 drives the TDM double buffer.
__device__ inline void gemm_core_tdm(const uint16_t* __restrict__ A,
                                     const uint16_t* __restrict__ B,
                                     uint16_t (*lA)[128 * 32],
                                     uint16_t (*lB)[256 * 32],
                                     int K, int mblk, int nblk, int tid,
                                     v8f (&acc)[4][4]) {
    const int lane  = tid & 31;
    const int wid   = tid >> 5;
    const int mwave = (wid >> 2) * 64;
    const int nwave = (wid & 3) * 64;
    const int g     = lane >> 4;
    const int l     = lane & 15;

    if (wid == 0) {
        tdm_load_2d(lds_off_u32(lA[0]), A + (size_t)mblk * K + 0,  32, 128, K);
        tdm_load_2d(lds_off_u32(lB[0]), B + (size_t)nblk * K + 0,  32, 256, K);
        tdm_load_2d(lds_off_u32(lA[1]), A + (size_t)mblk * K + 32, 32, 128, K);
        tdm_load_2d(lds_off_u32(lB[1]), B + (size_t)nblk * K + 32, 32, 256, K);
    }

    for (int kb = 0; kb < K; kb += 32) {
        const int cur = (kb >> 5) & 1;
        if (wid == 0) {
            if (kb + 32 < K) __builtin_amdgcn_s_wait_tensorcnt(2);
            else             __builtin_amdgcn_s_wait_tensorcnt(0);
        }
        __syncthreads();

        AFrag a[4], b[4];
#pragma unroll
        for (int i = 0; i < 4; ++i) {
            const uint16_t* base = lA[cur] + (size_t)(mwave + i * 16 + l) * 32;
            a[i].q[0] = *(const v4u*)(base + 16 * g);
            a[i].q[1] = *(const v4u*)(base + 16 * g + 8);
        }
#pragma unroll
        for (int j = 0; j < 4; ++j) {
            const uint16_t* base =
                lB[cur] + (size_t)(nwave + j * 16 + l) * 32 + 16 * g;
            b[j].q[0] = *(const v4u*)(base);
            b[j].q[1] = *(const v4u*)(base + 8);
        }
#pragma unroll
        for (int i = 0; i < 4; ++i)
#pragma unroll
            for (int j = 0; j < 4; ++j)
                acc[i][j] = bfwmma(a[i], b[j], acc[i][j]);

        __syncthreads();   // everyone done reading buffer 'cur'
        if (wid == 0 && kb + 64 < K) {
            tdm_load_2d(lds_off_u32(lA[cur]), A + (size_t)mblk * K + (kb + 64),
                        32, 128, K);
            tdm_load_2d(lds_off_u32(lB[cur]), B + (size_t)nblk * K + (kb + 64),
                        32, 256, K);
        }
    }
}

// ----------------------------------------------------------- QKV GEMM
__global__ __launch_bounds__(256) void gemm_qkv(const uint16_t* __restrict__ xbf,
                                                const uint16_t* __restrict__ wbf,
                                                uint16_t* __restrict__ qb,
                                                uint16_t* __restrict__ kb,
                                                uint16_t* __restrict__ vT) {
    __shared__ __align__(16) uint16_t lA[2][128 * 32];
    __shared__ __align__(16) uint16_t lB[2][256 * 32];

    const int tid  = threadIdx.x;
    const int lane = tid & 31;
    const int wid  = tid >> 5;
    const int mblk = blockIdx.y * 128;
    const int nblk = blockIdx.x * 256;
    const int m0   = mblk + (wid >> 2) * 64;
    const int n0   = nblk + (wid & 3) * 64;

    v8f acc[4][4] = {};
    gemm_core_tdm(xbf, wbf, lA, lB, DIM, mblk, nblk, tid, acc);

    const int g = lane >> 4;
    const int l = lane & 15;
#pragma unroll
    for (int j = 0; j < 4; ++j) {
        const int n    = n0 + j * 16 + l;      // 0..3071
        const int sect = n >> 10;              // 0=q 1=k 2=v
        const int f    = n & 1023;
        const int h    = f >> 6;
        const int d    = f & 63;
#pragma unroll
        for (int i = 0; i < 4; ++i) {
            const int    mrow0 = m0 + i * 16 + 8 * g;   // 8 consecutive tokens
            const int    bidx  = mrow0 >> 11;
            const int    t0    = mrow0 & (SEQ - 1);
            const size_t headb = (size_t)(bidx * HEADS + h);
            if (sect == 2) {
                uint32_t pk[4];
#pragma unroll
                for (int r = 0; r < 4; ++r) {
                    uint32_t lo = f2bf_bits(acc[i][j][2 * r]);
                    uint32_t hi = f2bf_bits(acc[i][j][2 * r + 1]);
                    pk[r] = lo | (hi << 16);
                }
                v4u p4 = {pk[0], pk[1], pk[2], pk[3]};
                *(v4u*)(vT + (headb * HD + d) * SEQ + t0) = p4;  // 8 tokens, 16B
            } else {
                uint16_t*   dst = (sect == 0) ? qb : kb;
                const float s   = (sect == 0) ? SCALE : 1.0f;
                const size_t base = headb * SEQ * HD + d;
#pragma unroll
                for (int r = 0; r < 8; ++r)
                    dst[base + (size_t)(t0 + r) * HD] = f2bf_bits(acc[i][j][r] * s);
            }
        }
    }
}

// ------------------------------------------------------ output GEMM
__global__ __launch_bounds__(256) void gemm_out(const uint16_t* __restrict__ attn,
                                                const uint16_t* __restrict__ wo,
                                                const float* __restrict__ bias,
                                                float* __restrict__ out) {
    __shared__ __align__(16) uint16_t lA[2][128 * 32];
    __shared__ __align__(16) uint16_t lB[2][256 * 32];

    const int tid  = threadIdx.x;
    const int lane = tid & 31;
    const int wid  = tid >> 5;
    const int mblk = blockIdx.y * 128;
    const int nblk = blockIdx.x * 256;
    const int m0   = mblk + (wid >> 2) * 64;
    const int n0   = nblk + (wid & 3) * 64;

    v8f acc[4][4] = {};
    gemm_core_tdm(attn, wo, lA, lB, DIM, mblk, nblk, tid, acc);

    const int g = lane >> 4;
    const int l = lane & 15;
#pragma unroll
    for (int j = 0; j < 4; ++j) {
        const int   n  = n0 + j * 16 + l;
        const float bv = bias[n];
#pragma unroll
        for (int i = 0; i < 4; ++i) {
            const int mrow0 = m0 + i * 16 + 8 * g;
#pragma unroll
            for (int r = 0; r < 8; ++r)
                out[(size_t)(mrow0 + r) * DIM + n] = acc[i][j][r] + bv;
        }
    }
}

// ------------------------------------------------------ flash attention
// WG = 256 thr / 8 waves = 128 queries; one wave owns 16 queries. Keys stream
// in steps of 32 through TDM-staged LDS tiles (K: 32x64, V^T: 64x32), double
// buffered; per-wave P bounce through LDS for the PV matmul.
__global__ __launch_bounds__(256) void flash_attn(const uint16_t* __restrict__ qb,
                                                  const uint16_t* __restrict__ kbuf,
                                                  const uint16_t* __restrict__ vT,
                                                  uint16_t* __restrict__ attnout) {
    __shared__ __align__(16) uint16_t ktile[2][32 * 64];  // keys x d
    __shared__ __align__(16) uint16_t vtile[2][64 * 32];  // d x keys
    __shared__ __align__(16) uint16_t ptile[8][16 * 32];  // per-wave P

    const int lane = threadIdx.x & 31;
    const int wid  = threadIdx.x >> 5;
    const int g    = lane >> 4;
    const int l    = lane & 15;
    const int bh   = blockIdx.x;             // 0..63
    const int b    = bh >> 4;
    const int h    = bh & 15;
    const int q0   = blockIdx.y * 128 + wid * 16;

    const uint16_t* qhead = qb   + (size_t)bh * SEQ * HD;
    const uint16_t* khead = kbuf + (size_t)bh * SEQ * HD;
    const uint16_t* vhead = vT   + (size_t)bh * HD * SEQ;

    // Q A-fragments (16x64 => two 16x32 frags), kept in registers.
    AFrag qa[2];
    {
        const uint32_t* qrow = (const uint32_t*)(qhead + (size_t)(q0 + l) * HD);
#pragma unroll
        for (int f = 0; f < 2; ++f)
#pragma unroll
            for (int v = 0; v < 8; ++v) {
                int k = f * 32 + 16 * g + 2 * v;
                qa[f].u[v] = qrow[k >> 1];
            }
    }

    if (wid == 0) {
        tdm_load_2d(lds_off_u32(ktile[0]), khead,                 64, 32, HD);
        tdm_load_2d(lds_off_u32(vtile[0]), vhead,                 32, 64, SEQ);
        tdm_load_2d(lds_off_u32(ktile[1]), khead + (size_t)32 * HD, 64, 32, HD);
        tdm_load_2d(lds_off_u32(vtile[1]), vhead + 32,            32, 64, SEQ);
    }

    v8f   o[4] = {};
    float mrun[8], lrun[8];
#pragma unroll
    for (int r = 0; r < 8; ++r) { mrun[r] = -1e30f; lrun[r] = 0.0f; }

    for (int j = 0; j < SEQ; j += 32) {
        const int cur = (j >> 5) & 1;
        if (wid == 0) {
            if (j + 32 < SEQ) __builtin_amdgcn_s_wait_tensorcnt(2);
            else              __builtin_amdgcn_s_wait_tensorcnt(0);
        }
        __syncthreads();

        // S tiles: S[st] = Q (16x64) * K[st*16..+16]^T from LDS (2 k-steps).
        v8f s[2];
#pragma unroll
        for (int st = 0; st < 2; ++st) {
            v8f acc = {};
            const uint16_t* krow = ktile[cur] + (size_t)(st * 16 + l) * 64;
#pragma unroll
            for (int f = 0; f < 2; ++f) {
                AFrag kf;
                kf.q[0] = *(const v4u*)(krow + f * 32 + 16 * g);
                kf.q[1] = *(const v4u*)(krow + f * 32 + 16 * g + 8);
                acc = bfwmma(qa[f], kf, acc);
            }
            s[st] = acc;
        }

        // Online softmax (rows r+8g; reduce across 16 lanes holding columns).
        float p0v[8], p1v[8];
#pragma unroll
        for (int r = 0; r < 8; ++r) {
            float mx = fmaxf(s[0][r], s[1][r]);
#pragma unroll
            for (int w = 1; w < 16; w <<= 1) mx = fmaxf(mx, __shfl_xor(mx, w, 16));
            float mnew  = fmaxf(mrun[r], mx);
            float alpha = __expf(mrun[r] - mnew);
            float p0 = __expf(s[0][r] - mnew);
            float p1 = __expf(s[1][r] - mnew);
            float rs = p0 + p1;
#pragma unroll
            for (int w = 1; w < 16; w <<= 1) rs += __shfl_xor(rs, w, 16);
            lrun[r] = lrun[r] * alpha + rs;
            mrun[r] = mnew;
            p0v[r] = p0; p1v[r] = p1;
#pragma unroll
            for (int dt = 0; dt < 4; ++dt) o[dt][r] = o[dt][r] * alpha;
        }

        // Bounce P (C-layout) through per-wave LDS to A-fragment layout.
        uint16_t* pt = ptile[wid];
#pragma unroll
        for (int r = 0; r < 8; ++r) {
            pt[(r + 8 * g) * 32 + l]      = f2bf_bits(p0v[r]);
            pt[(r + 8 * g) * 32 + 16 + l] = f2bf_bits(p1v[r]);
        }
        asm volatile("s_wait_dscnt 0" ::: "memory");   // CDNA5 split DS counter
        AFrag pa;
#pragma unroll
        for (int v = 0; v < 8; ++v) {
            int k = 16 * g + 2 * v;
            pa.u[v] = *(const uint32_t*)(pt + (size_t)l * 32 + k);
        }

        // O += P (16x32) * V (32x64) from the staged V^T tile.
#pragma unroll
        for (int dt = 0; dt < 4; ++dt) {
            AFrag vf;
            const uint16_t* vrow = vtile[cur] + (size_t)(dt * 16 + l) * 32 + 16 * g;
            vf.q[0] = *(const v4u*)(vrow);
            vf.q[1] = *(const v4u*)(vrow + 8);
            o[dt] = bfwmma(pa, vf, o[dt]);
        }

        __syncthreads();   // everyone done reading buffer 'cur'
        if (wid == 0 && j + 64 < SEQ) {
            tdm_load_2d(lds_off_u32(ktile[cur]),
                        khead + (size_t)(j + 64) * HD, 64, 32, HD);
            tdm_load_2d(lds_off_u32(vtile[cur]),
                        vhead + (j + 64),              32, 64, SEQ);
        }
    }

    // Normalize and store [b, t, h*64 + d] as bf16.
#pragma unroll
    for (int r = 0; r < 8; ++r) {
        float inv = 1.0f / lrun[r];
        int   t   = q0 + r + 8 * g;
        size_t rowb = ((size_t)b * SEQ + t) * DIM + h * HD;
#pragma unroll
        for (int dt = 0; dt < 4; ++dt)
            attnout[rowb + dt * 16 + l] = f2bf_bits(o[dt][r] * inv);
    }
}

// ------------------------------------------------------------- launch
extern "C" void kernel_launch(void* const* d_in, const int* in_sizes, int n_in,
                              void* d_out, int out_size, void* d_ws, size_t ws_size,
                              hipStream_t stream) {
    (void)in_sizes; (void)n_in; (void)out_size; (void)ws_size;

    const float* x     = (const float*)d_in[0];
    const float* w_qkv = (const float*)d_in[1];
    const float* w_out = (const float*)d_in[2];
    const float* b_out = (const float*)d_in[3];
    float*       out   = (float*)d_out;

    // Workspace layout (bf16, ~88 MB total).
    uint16_t* p      = (uint16_t*)d_ws;
    uint16_t* xbf    = p; p += (size_t)NTOK * DIM;      // 8M
    uint16_t* wqkvbf = p; p += (size_t)3 * DIM * DIM;   // 3M
    uint16_t* woutbf = p; p += (size_t)DIM * DIM;       // 1M
    uint16_t* qb     = p; p += (size_t)NTOK * DIM;      // [b,h,n,d]
    uint16_t* kb     = p; p += (size_t)NTOK * DIM;      // [b,h,n,d]
    uint16_t* vTb    = p; p += (size_t)NTOK * DIM;      // [b,h,d,n]
    uint16_t* attnbf = p;                               // [b,n,dim]

    const int thr = 256;
    cvt_f32_bf16<<<(NTOK * DIM + thr - 1) / thr, thr, 0, stream>>>(x, xbf, NTOK * DIM);
    cvt_f32_bf16<<<(3 * DIM * DIM + thr - 1) / thr, thr, 0, stream>>>(w_qkv, wqkvbf, 3 * DIM * DIM);
    cvt_f32_bf16<<<(DIM * DIM + thr - 1) / thr, thr, 0, stream>>>(w_out, woutbf, DIM * DIM);

    gemm_qkv<<<dim3(3072 / 256, NTOK / 128), dim3(256), 0, stream>>>(xbf, wqkvbf, qb, kb, vTb);

    flash_attn<<<dim3(BATCH * HEADS, SEQ / 128), dim3(256), 0, stream>>>(qb, kb, vTb, attnbf);

    gemm_out<<<dim3(DIM / 256, NTOK / 128), dim3(256), 0, stream>>>(attnbf, woutbf, b_out, out);
}
